// SAKT_29901562315289
// MI455X (gfx1250) — matrix operand order repaired and checked by reading, not compiled, not run on hardware
//
#include <hip/hip_runtime.h>
#include <hip/hip_bf16.h>

// ---------------- problem constants (match reference) ----------------
constexpr int Bc = 32, Sc = 1024, Ec = 512, Hc = 8, Lc = 2, Dc = 64;
constexpr int Mtok = Bc * Sc; // 32768 tokens

typedef __bf16 bf16;
typedef __attribute__((ext_vector_type(16))) __bf16 v16bf;
typedef __attribute__((ext_vector_type(8)))  __bf16 v8bf;
typedef __attribute__((ext_vector_type(8)))  float  v8f;

#define USE_ASYNC_LDS 1

__device__ __forceinline__ v8f wmma_bf16(v16bf a, v16bf b, v8f c) {
  return __builtin_amdgcn_wmma_f32_16x16x32_bf16(
      /*neg_a=*/false, a, /*neg_b=*/false, b,
      /*c_mod=*/(short)0, c, /*reuse_a=*/false, /*reuse_b=*/false);
}

// Two contiguous 16B halves -> one 32B fragment, kept in VGPRs (no alloca).
__device__ __forceinline__ v16bf frag2(const bf16* lo, const bf16* hi) {
  v8bf a = *(const v8bf*)lo;
  v8bf b = *(const v8bf*)hi;
  return __builtin_shufflevector(a, b, 0, 1, 2, 3, 4, 5, 6, 7,
                                       8, 9, 10, 11, 12, 13, 14, 15);
}

#if USE_ASYNC_LDS
// CDNA5 async copy: global -> LDS, 16B per lane, tracked by ASYNCcnt.
__device__ __forceinline__ void async_b128(const void* gptr, const void* lptr) {
  unsigned loff = (unsigned)(uintptr_t)lptr;           // low 32 bits = LDS byte addr
  unsigned long long g = (unsigned long long)(uintptr_t)gptr;
  asm volatile("global_load_async_to_lds_b128 %0, %1, off"
               :: "v"(loff), "v"(g) : "memory");
}
// Wait until at most `n` async ops outstanding (loads complete in order).
#define WAIT_ASYNC(n) asm volatile("s_wait_asynccnt " #n ::: "memory")
#else
#define WAIT_ASYNC(n)
#endif

// ---------------- weight prep: W[K=E][N=E] f32 -> Wt[N][K] bf16 ----------------
__global__ __launch_bounds__(256) void prep_wt(const float* __restrict__ W,
                                               bf16* __restrict__ Wt) {
  int i = blockIdx.x * 256 + threadIdx.x;     // i = k*E + n
  int n = i % Ec, k = i / Ec;
  Wt[(size_t)n * Ec + k] = (bf16)W[i];
}

// ---------------- embedding + positional encoding ----------------
__global__ __launch_bounds__(256) void embed_pe(const int* __restrict__ inp,
                                                const float* __restrict__ table,
                                                float* __restrict__ x32,
                                                bf16* __restrict__ xbf) {
  size_t i = (size_t)blockIdx.x * 256 + threadIdx.x;  // over M*E
  int e = (int)(i % Ec);
  size_t t = i / Ec;                // token index b*S+s
  int s = (int)(t % Sc);
  int tok = inp[t];
  float div = __expf((float)(e & ~1) * (-9.210340371976184f / (float)Ec));
  float ang = (float)s * div;
  float pe = (e & 1) ? __cosf(ang) : __sinf(ang);
  float v = table[(size_t)tok * Ec + e] + pe;
  x32[i] = v;
  xbf[i] = (bf16)v;
}

// ---------------- bf16 GEMM:  C[M,N] = A[M,K] @ Wt[N,K]^T + bias ----------------
// Double-buffered async-LDS pipeline; 8 WMMAs per 32-K step per wave.
// mode 0: Cbf = bf16(C)                      (Q/K/V projections)
// mode 1: x = x + relu(C); write x32 + xbf   (O-proj + residual)
__global__ __launch_bounds__(256) void gemm_bf16(const bf16* __restrict__ A,
                                                 const bf16* __restrict__ Wt,
                                                 const float* __restrict__ bias,
                                                 bf16* __restrict__ Cbf,
                                                 float* __restrict__ Xf,
                                                 bf16* __restrict__ Xbf,
                                                 int M, int N, int K, int mode) {
  constexpr int BM = 128, BN = 128, BK = 32;
  __shared__ bf16 sA[2][BM * BK];
  __shared__ bf16 sB[2][BN * BK];
  const int tid = threadIdx.x;
  const int lane = tid & 31;
  const int wave = tid >> 5;            // 8 waves
  const int wm = wave & 1;              // 2 waves along M (64 rows each)
  const int wn = wave >> 1;             // 4 waves along N (32 cols each)
  const int mblk = M / BM;
  const int bm = (blockIdx.x % mblk) * BM;
  const int bn = (blockIdx.x / mblk) * BN;

  v8f acc[4][2] = {};
  const int arow = wm * 64 + (lane & 15);
  const int kh = (lane >> 4) * 8;       // A-fragment K half
  const int koff = (lane >> 4) * 8;     // B-fragment K half
  const int srow = tid >> 1, shalf = tid & 1;   // staging: 2 threads/row, 32B each

  auto stage = [&](int k0, int buf) {
    const bf16* ga = A  + (size_t)(bm + srow) * K + k0 + shalf * 16;
    const bf16* gb = Wt + (size_t)(bn + srow) * K + k0 + shalf * 16;
    bf16* la = sA[buf] + srow * BK + shalf * 16;
    bf16* lb = sB[buf] + srow * BK + shalf * 16;
#if USE_ASYNC_LDS
    async_b128(ga,     la);
    async_b128(ga + 8, la + 8);
    async_b128(gb,     lb);
    async_b128(gb + 8, lb + 8);
#else
    ((uint4*)la)[0] = ((const uint4*)ga)[0];
    ((uint4*)la)[1] = ((const uint4*)ga)[1];
    ((uint4*)lb)[0] = ((const uint4*)gb)[0];
    ((uint4*)lb)[1] = ((const uint4*)gb)[1];
#endif
  };

  stage(0, 0);                       // prologue: tile 0 in flight
  int buf = 0;
  for (int k0 = 0; k0 < K; k0 += BK, buf ^= 1) {
    const bool pre = (k0 + BK) < K;
    if (pre) {
      stage(k0 + BK, buf ^ 1);       // issue next tile before waiting
      WAIT_ASYNC(0x4);               // current tile done; next (4 ops) in flight
    } else {
      WAIT_ASYNC(0x0);
    }
    __syncthreads();

    v16bf afr[4], bfr[2];
#pragma unroll
    for (int i = 0; i < 4; i++) {
      const bf16* p = sA[buf] + (arow + i * 16) * BK;
      afr[i] = frag2(p + kh, p + 16 + kh);
    }
#pragma unroll
    for (int i = 0; i < 2; i++) {
      const bf16* p = sB[buf] + (wn * 32 + i * 16 + (lane & 15)) * BK + koff;
      bfr[i] = frag2(p, p + 16);
    }
#pragma unroll
    for (int i = 0; i < 4; i++)
#pragma unroll
      for (int j = 0; j < 2; j++)
        acc[i][j] = wmma_bf16(afr[i], bfr[j], acc[i][j]);
    __syncthreads();                 // all reads of this buffer retired
  }

  // epilogue (C layout: col = lane%16, row = r + (lane/16)*8)
  const int rl = lane >> 4, cl = lane & 15;
#pragma unroll
  for (int j = 0; j < 2; j++) {
    int col = bn + wn * 32 + j * 16 + cl;
    float bv = bias[col % N];
#pragma unroll
    for (int i = 0; i < 4; i++) {
      int row0 = bm + wm * 64 + i * 16 + rl * 8;
#pragma unroll
      for (int r = 0; r < 8; r++) {
        size_t idx = (size_t)(row0 + r) * N + col;
        float v = acc[i][j][r] + bv;
        if (mode == 0) {
          Cbf[idx] = (bf16)v;
        } else {
          float x = Xf[idx] + (v > 0.f ? v : 0.f);
          Xf[idx] = x;
          Xbf[idx] = (bf16)x;
        }
      }
    }
  }
}

// ---------------- causal flash attention (bf16 in/out, f32 softmax) ----------------
// grid: B*H*(S/128) blocks of 256 threads; wave w owns q rows [q0+16w, q0+16w+15].
// Double-buffered async staging of K/V tiles.
__global__ __launch_bounds__(256) void attn_flash(const bf16* __restrict__ Q,
                                                  const bf16* __restrict__ Kb,
                                                  const bf16* __restrict__ Vb,
                                                  bf16* __restrict__ O) {
  __shared__ bf16 sK[2][32 * 64];   // 32 keys x 64 dims (row = key)
  __shared__ bf16 sVr[2][32 * 64];  // raw V tile (row = key)
  __shared__ bf16 sVt[64 * 32];     // transposed: [d][key]
  __shared__ bf16 sP[8 * 16 * 32];  // per-wave P tile (16 q x 32 keys)

  const int tid = threadIdx.x, lane = tid & 31, wave = tid >> 5;
  const int qblocks = Sc / 128;
  const int qblk = blockIdx.x % qblocks;
  const int h = (blockIdx.x / qblocks) % Hc;
  const int b = blockIdx.x / (qblocks * Hc);
  const int q0 = qblk * 128;
  const int qr0 = q0 + wave * 16;

  const int rl = lane >> 4, cl = lane & 15;
  const int kh = rl * 8;       // A-fragment K half
  const int koff = rl * 8;     // B-fragment K half

  // Q fragments for this wave's 16 rows (two k=32 fragments over D=64)
  v16bf qf[2];
  {
    const bf16* qp = Q + ((size_t)(b * Sc + qr0 + cl)) * Ec + h * Dc;
    qf[0] = frag2(qp + kh,      qp + 16 + kh);
    qf[1] = frag2(qp + 32 + kh, qp + 48 + kh);
  }

  v8f o[4] = {};
  float m[8], l[8];
#pragma unroll
  for (int r = 0; r < 8; r++) { m[r] = -1e30f; l[r] = 0.f; }

  const int skey = tid >> 3, schunk = tid & 7;  // staging: 16B per thread per tensor

  auto stageKV = [&](int kb, int buf) {
    const bf16* gk = Kb + ((size_t)(b * Sc + kb + skey)) * Ec + h * Dc + schunk * 8;
    const bf16* gv = Vb + ((size_t)(b * Sc + kb + skey)) * Ec + h * Dc + schunk * 8;
#if USE_ASYNC_LDS
    async_b128(gk, sK[buf]  + skey * 64 + schunk * 8);
    async_b128(gv, sVr[buf] + skey * 64 + schunk * 8);
#else
    ((uint4*)(sK[buf]  + skey * 64 + schunk * 8))[0] = ((const uint4*)gk)[0];
    ((uint4*)(sVr[buf] + skey * 64 + schunk * 8))[0] = ((const uint4*)gv)[0];
#endif
  };

  const int kend = q0 + 128;
  stageKV(0, 0);
  int buf = 0;
  for (int kb = 0; kb < kend; kb += 32, buf ^= 1) {
    const bool pre = (kb + 32) < kend;
    if (pre) {
      stageKV(kb + 32, buf ^ 1);     // next K/V tile in flight behind compute
      WAIT_ASYNC(0x2);
    } else {
      WAIT_ASYNC(0x0);
    }
    __syncthreads();                 // current K/V tile visible to all waves

    // transpose V inside LDS: sVt[d][key] = sVr[buf][key][d]   (8 elems/thread)
#pragma unroll
    for (int i = 0; i < 8; i++) {
      int idx = tid * 8 + i, key = idx >> 6, d = idx & 63;
      sVt[d * 32 + key] = sVr[buf][key * 64 + d];
    }

    const bool active = (kb <= qr0 + 15);
    v16bf pf;
    if (active) {
      // scores: two 16x16 tiles (keys kb..+15, kb+16..+31), K-dim = D in two steps
      v8f s0 = {}, s1 = {};
#pragma unroll
      for (int step = 0; step < 2; step++) {
        const bf16* p0 = sK[buf] + cl * 64 + step * 32 + koff;
        const bf16* p1 = sK[buf] + (16 + cl) * 64 + step * 32 + koff;
        v16bf bk0 = frag2(p0, p0 + 16);
        v16bf bk1 = frag2(p1, p1 + 16);
        s0 = wmma_bf16(qf[step], bk0, s0);
        s1 = wmma_bf16(qf[step], bk1, s1);
      }
      // online softmax per row; write P (bf16) into wave-private LDS
      const float scale = 0.125f;   // 1/sqrt(64)
      bf16* pbase = sP + wave * (16 * 32);
#pragma unroll
      for (int r = 0; r < 8; r++) {
        int qrow = qr0 + r + rl * 8;
        float v0 = ((kb + cl)      <= qrow) ? s0[r] * scale : -1e30f;
        float v1 = ((kb + 16 + cl) <= qrow) ? s1[r] * scale : -1e30f;
        float mx = fmaxf(v0, v1);
#pragma unroll
        for (int off = 8; off; off >>= 1) mx = fmaxf(mx, __shfl_xor(mx, off, 32));
        float mnew = fmaxf(m[r], mx);
        float fac = __expf(m[r] - mnew);
        float p0 = __expf(v0 - mnew), p1 = __expf(v1 - mnew);
        float rs = p0 + p1;
#pragma unroll
        for (int off = 8; off; off >>= 1) rs += __shfl_xor(rs, off, 32);
        l[r] = l[r] * fac + rs;
        m[r] = mnew;
#pragma unroll
        for (int t = 0; t < 4; t++) o[t][r] *= fac;
        bf16* pp = pbase + (r + rl * 8) * 32;
        pp[cl] = (bf16)p0;
        pp[16 + cl] = (bf16)p1;
      }
      // DS ops are in-order within a wave; make the LDS transpose hand-off explicit
      asm volatile("s_wait_dscnt 0" ::: "memory");
      const bf16* p = pbase + cl * 32;
      pf = frag2(p + kh, p + 16 + kh);
    }
    __syncthreads();   // sVt complete (all threads' transpose writes visible)
    if (active) {
#pragma unroll
      for (int t = 0; t < 4; t++) {
        const bf16* p = sVt + (t * 16 + cl) * 32 + koff;
        v16bf bv = frag2(p, p + 16);
        o[t] = wmma_bf16(pf, bv, o[t]);
      }
    }
    __syncthreads();   // readers of sK/sVt done before next refill/transpose
  }

  // normalize and write this head's slice
#pragma unroll
  for (int r = 0; r < 8; r++) {
    float inv = l[r] > 0.f ? 1.f / l[r] : 0.f;
    int qrow = qr0 + r + rl * 8;
    bf16* op = O + ((size_t)(b * Sc + qrow)) * Ec + h * Dc;
#pragma unroll
    for (int t = 0; t < 4; t++) op[t * 16 + cl] = (bf16)(o[t][r] * inv);
  }
}

// ---------------- final head: out[t] = x[t,:] . Wout + bout ----------------
__global__ __launch_bounds__(256) void final_proj(const float* __restrict__ x32,
                                                  const float* __restrict__ Wout,
                                                  const float* __restrict__ bout,
                                                  float* __restrict__ out) {
  int token = blockIdx.x * 8 + (threadIdx.x >> 5);
  int lane = threadIdx.x & 31;
  const float* xp = x32 + (size_t)token * Ec;
  float sum = 0.f;
#pragma unroll
  for (int j = 0; j < Ec / 32; j++) sum += xp[lane + j * 32] * Wout[lane + j * 32];
#pragma unroll
  for (int off = 16; off; off >>= 1) sum += __shfl_xor(sum, off, 32);
  if (lane == 0) out[token] = sum + bout[0];
}

// ---------------- host orchestration ----------------
extern "C" void kernel_launch(void* const* d_in, const int* in_sizes, int n_in,
                              void* d_out, int out_size, void* d_ws, size_t ws_size,
                              hipStream_t stream) {
  const int*   inputs = (const int*)d_in[0];
  const float* table  = (const float*)d_in[1];
  const float* Wq = (const float*)d_in[2];  const float* bq = (const float*)d_in[3];
  const float* Wk = (const float*)d_in[4];  const float* bk = (const float*)d_in[5];
  const float* Wv = (const float*)d_in[6];  const float* bv = (const float*)d_in[7];
  const float* Wo = (const float*)d_in[8];  const float* bo = (const float*)d_in[9];
  const float* Wout = (const float*)d_in[10];
  const float* bout = (const float*)d_in[11];
  float* out = (float*)d_out;

  // workspace carve-out
  char* ws = (char*)d_ws;
  const size_t actF = (size_t)Mtok * Ec;            // elements per activation
  float* x32 = (float*)ws;           ws += actF * sizeof(float);
  bf16*  xbf = (bf16*)ws;            ws += actF * sizeof(bf16);
  bf16*  qbf = (bf16*)ws;            ws += actF * sizeof(bf16);
  bf16*  kbf = (bf16*)ws;            ws += actF * sizeof(bf16);
  bf16*  vbf = (bf16*)ws;            ws += actF * sizeof(bf16);
  bf16*  abf = (bf16*)ws;            ws += actF * sizeof(bf16);
  bf16*  wt  = (bf16*)ws;            // [L][4][E][E] transposed bf16 weights

  const size_t wsz = (size_t)Ec * Ec;
  const float* Wsrc[4] = {Wq, Wk, Wv, Wo};
  for (int l = 0; l < Lc; l++)
    for (int w = 0; w < 4; w++)
      prep_wt<<<(int)(wsz / 256), 256, 0, stream>>>(Wsrc[w] + (size_t)l * wsz,
                                                    wt + ((size_t)(l * 4 + w)) * wsz);

  embed_pe<<<(int)(actF / 256), 256, 0, stream>>>(inputs, table, x32, xbf);

  const int gemm_grid = (Mtok / 128) * (Ec / 128);  // 1024
  const int attn_grid = Bc * Hc * (Sc / 128);       // 2048

  for (int l = 0; l < Lc; l++) {
    bf16* wtq = wt + ((size_t)(l * 4 + 0)) * wsz;
    bf16* wtk = wt + ((size_t)(l * 4 + 1)) * wsz;
    bf16* wtv = wt + ((size_t)(l * 4 + 2)) * wsz;
    bf16* wto = wt + ((size_t)(l * 4 + 3)) * wsz;
    gemm_bf16<<<gemm_grid, 256, 0, stream>>>(xbf, wtq, bq + (size_t)l * Ec,
                                             qbf, nullptr, nullptr, Mtok, Ec, Ec, 0);
    gemm_bf16<<<gemm_grid, 256, 0, stream>>>(xbf, wtk, bk + (size_t)l * Ec,
                                             kbf, nullptr, nullptr, Mtok, Ec, Ec, 0);
    gemm_bf16<<<gemm_grid, 256, 0, stream>>>(xbf, wtv, bv + (size_t)l * Ec,
                                             vbf, nullptr, nullptr, Mtok, Ec, Ec, 0);
    attn_flash<<<attn_grid, 256, 0, stream>>>(qbf, kbf, vbf, abf);
    gemm_bf16<<<gemm_grid, 256, 0, stream>>>(abf, wto, bo + (size_t)l * Ec,
                                             nullptr, x32, xbf, Mtok, Ec, Ec, 1);
  }

  final_proj<<<Mtok / 8, 256, 0, stream>>>(x32, Wout, bout, out);
  (void)in_sizes; (void)n_in; (void)out_size; (void)ws_size;
}
